// MinimalRNNCell_87522843558638
// MI455X (gfx1250) — matrix operand order repaired and checked by reading, not compiled, not run on hardware
//
#include <hip/hip_runtime.h>
#include <hip/hip_bf16.h>

typedef __attribute__((ext_vector_type(16))) __bf16 v16bf;
typedef __attribute__((ext_vector_type(8)))  float  v8f;

#define B_  32
#define T_  2048
#define D_  1024
#define U_  1024
#define NWG2 64          // workgroups in the persistent scan kernel
#define LDS_STRIDE 1032  // 1024 + 8 bf16 pad (16B) to spread LDS banks

// ---------- helpers ----------
static __device__ __forceinline__ unsigned short f2bf_bits(float f) {
    union { float f; unsigned u; } c; c.f = f;
    unsigned u = c.u;
    // round-to-nearest-even f32 -> bf16
    unsigned r = (u + 0x7FFFu + ((u >> 16) & 1u)) >> 16;
    return (unsigned short)r;
}

static __device__ __forceinline__ v8f wmma_bf16(const v16bf& a, const v16bf& b, v8f c) {
    return __builtin_amdgcn_wmma_f32_16x16x32_bf16(false, a, false, b, (short)0, c,
                                                   false, false);
}

// ---------- weight transpose + f32->bf16 ----------
// Kt[u*D + d] = kernel[d*U + u], Rt[u*U + k] = R[k*U + u]
__global__ void convert_weights(const float* __restrict__ kern,
                                const float* __restrict__ rker,
                                unsigned short* __restrict__ Kt,
                                unsigned short* __restrict__ Rt) {
    int idx = blockIdx.x * blockDim.x + threadIdx.x;   // over D_*U_
    int d = idx / U_;
    int u = idx % U_;
    Kt[(size_t)u * D_ + d] = f2bf_bits(kern[idx]);
    Rt[(size_t)u * U_ + d] = f2bf_bits(rker[idx]);
}

// ---------- h0 -> bf16 hbuf[0], reset barrier counter ----------
__global__ void init_h(const float* __restrict__ h0,
                       unsigned short* __restrict__ hbuf,
                       unsigned* __restrict__ counter) {
    int i = blockIdx.x * blockDim.x + threadIdx.x;
    if (i < B_ * U_) hbuf[i] = f2bf_bits(h0[i]);
    if (i == 0) *counter = 0u;
}

// ---------- x f32 -> bf16 (fast path) ----------
__global__ void convert_x(const float* __restrict__ x,
                          unsigned short* __restrict__ xbf) {
    size_t i = (size_t)blockIdx.x * blockDim.x + threadIdx.x;  // over (B*T*D)/4
    float4 f = ((const float4*)x)[i];
    unsigned v0 = (unsigned)f2bf_bits(f.x) | ((unsigned)f2bf_bits(f.y) << 16);
    unsigned v1 = (unsigned)f2bf_bits(f.z) | ((unsigned)f2bf_bits(f.w) << 16);
    uint2 o; o.x = v0; o.y = v1;
    ((uint2*)xbf)[i] = o;
}

// ---------- phase 1: xw = x @ kernel + const  -> d_out ----------
// grid (M/128, N/64), 128 threads = 4 waves stacked on M; each wave: 32x64 tile
// (two 16-row subtiles share every B fetch -> 1.5 b128 loads per WMMA per lane).
template<bool PRE>
__global__ void __launch_bounds__(128)
xw_gemm(const float* __restrict__ x,
        const unsigned short* __restrict__ xbf,
        const unsigned short* __restrict__ Kt,
        const float* __restrict__ cst,
        float* __restrict__ out) {
    const int wave = threadIdx.x >> 5;
    const int lane = threadIdx.x & 31;
    const int r    = lane & 15;
    const int hi   = lane >> 4;          // 0/1
    const int tileM = blockIdx.x * 128 + wave * 32;
    const int cbase = blockIdx.y * 64;
    const int kh    = hi * 8;            // A half-select (k 0-7 / 8-15, +16 run)
    const int khb   = hi * 16;           // unused; B is loaded with kh, same k order as A

    v8f acc[2][4] = {};

    for (int kk = 0; kk < D_; kk += 32) {
        union { uint4 q[2]; v16bf v; } A[2];
        #pragma unroll
        for (int mi = 0; mi < 2; ++mi) {
            const int row = tileM + mi * 16 + r;
            if (PRE) {
                const unsigned short* ap = xbf + (size_t)row * D_ + kk + kh;
                A[mi].q[0] = *(const uint4*)(ap);
                A[mi].q[1] = *(const uint4*)(ap + 16);
            } else {
                const float4* ap = (const float4*)(x + (size_t)row * D_ + kk + kh);
                float4 f0 = ap[0], f1 = ap[1], f2 = ap[4], f3 = ap[5];
                unsigned short* s = (unsigned short*)&A[mi];
                s[0]=f2bf_bits(f0.x); s[1]=f2bf_bits(f0.y); s[2]=f2bf_bits(f0.z); s[3]=f2bf_bits(f0.w);
                s[4]=f2bf_bits(f1.x); s[5]=f2bf_bits(f1.y); s[6]=f2bf_bits(f1.z); s[7]=f2bf_bits(f1.w);
                s[8]=f2bf_bits(f2.x); s[9]=f2bf_bits(f2.y); s[10]=f2bf_bits(f2.z); s[11]=f2bf_bits(f2.w);
                s[12]=f2bf_bits(f3.x); s[13]=f2bf_bits(f3.y); s[14]=f2bf_bits(f3.z); s[15]=f2bf_bits(f3.w);
            }
        }
        #pragma unroll
        for (int j = 0; j < 4; ++j) {
            const int col = cbase + j * 16 + r;
            union { uint4 q[2]; v16bf v; } Bv;
            const unsigned short* bp = Kt + (size_t)col * D_ + kk + kh;
            Bv.q[0] = *(const uint4*)(bp);
            Bv.q[1] = *(const uint4*)(bp + 16);
            acc[0][j] = wmma_bf16(A[0].v, Bv.v, acc[0][j]);
            acc[1][j] = wmma_bf16(A[1].v, Bv.v, acc[1][j]);
        }
    }

    const int b = (blockIdx.x * 128) / T_;   // 128-row block never crosses batch boundary
    #pragma unroll
    for (int mi = 0; mi < 2; ++mi) {
        #pragma unroll
        for (int j = 0; j < 4; ++j) {
            const int col = cbase + j * 16 + r;
            const float cv = cst[(size_t)b * U_ + col];
            #pragma unroll
            for (int v = 0; v < 8; ++v) {
                const int m = tileM + mi * 16 + hi * 8 + v;
                out[(size_t)m * U_ + col] = acc[mi][j][v] + cv;
            }
        }
    }
}

// ---------- phase 2: persistent recurrent scan ----------
// 64 WGs x 64 threads (2 waves). WG g owns columns [g*16, g*16+16).
// Wave w computes rows [16w, 16w+16). R-slice resident in LDS for all T steps,
// staged once with CDNA5 async global->LDS copies (ASYNCcnt).
__global__ void __launch_bounds__(64)
scan_kernel(float* __restrict__ out,
            const unsigned short* __restrict__ Rt,   // Rt[u*U + k]
            unsigned short* __restrict__ hbuf,       // [2][B_*U_] bf16
            unsigned* __restrict__ counter) {
    __shared__ unsigned short ldsR[16 * LDS_STRIDE];

    const int g    = blockIdx.x;
    const int tid  = threadIdx.x;
    const int wave = tid >> 5;
    const int lane = tid & 31;
    const int r    = lane & 15;
    const int hi   = lane >> 4;
    const int kh   = hi * 8;     // A half
    const int khb  = hi * 16;    // unused; B also uses kh

    // stage this WG's 16 columns of R into LDS, read from HBM exactly once,
    // via async global->LDS (no VGPR round-trip, tracked with ASYNCcnt)
    for (int i = tid; i < 16 * 128; i += 64) {       // 128 uint4 per row
        const int n = i >> 7, kq = i & 127;
        unsigned long long ga =
            (unsigned long long)(const void*)(Rt + (size_t)(g * 16 + n) * U_ + kq * 8);
        unsigned lds_off = (unsigned)(uintptr_t)&ldsR[n * LDS_STRIDE + kq * 8];
        asm volatile("global_load_async_to_lds_b128 %0, %1, off"
                     :: "v"(lds_off), "v"(ga) : "memory");
    }
    asm volatile("s_wait_asynccnt 0" ::: "memory");
    __syncthreads();

    const int arow = wave * 16 + r;                  // batch row for A operand
    const int colg = g * 16 + r;                     // global output column
    const unsigned short* brow = &ldsR[r * LDS_STRIDE];

    for (int t = 0; t < T_; ++t) {
        const unsigned short* hp = hbuf + (size_t)(t & 1) * (B_ * U_);
        unsigned short*       hn = hbuf + (size_t)((t + 1) & 1) * (B_ * U_);

        // prefetch xw_t early so its L2 latency hides under the WMMA chain
        float xw[8];
        size_t oaddr[8];
        #pragma unroll
        for (int v = 0; v < 8; ++v) {
            const int b = wave * 16 + hi * 8 + v;
            oaddr[v] = ((size_t)b * T_ + t) * U_ + colg;
            xw[v] = out[oaddr[v]];
        }

        v8f acc = {};
        #pragma unroll 4
        for (int kk = 0; kk < U_; kk += 32) {
            union { uint4 q[2]; v16bf v; } A, Bv;
            const unsigned short* ap = hp + (size_t)arow * U_ + kk + kh;
            A.q[0] = *(const uint4*)(ap);
            A.q[1] = *(const uint4*)(ap + 16);
            const unsigned short* bp = brow + kk + kh;
            Bv.q[0] = *(const uint4*)(bp);
            Bv.q[1] = *(const uint4*)(bp + 16);
            acc = wmma_bf16(A.v, Bv.v, acc);
        }

        // h_t = xw_t + h_{t-1} @ R : in-place update of d_out, bf16 copy for peers
        #pragma unroll
        for (int v = 0; v < 8; ++v) {
            const int b = wave * 16 + hi * 8 + v;
            const float hv = acc[v] + xw[v];
            out[oaddr[v]] = hv;
            hn[(size_t)b * U_ + colg] = f2bf_bits(hv);
        }

        // device-wide step barrier
        __threadfence();                 // agent-scope release of our h stores
        __syncthreads();
        if (tid == 0) {
            atomicAdd(counter, 1u);
            const unsigned target = (unsigned)NWG2 * (unsigned)(t + 1);
            while (__hip_atomic_load(counter, __ATOMIC_RELAXED,
                                     __HIP_MEMORY_SCOPE_AGENT) < target) {
                __builtin_amdgcn_s_sleep(1);
            }
        }
        __syncthreads();
        __builtin_amdgcn_fence(__ATOMIC_ACQUIRE, "agent");  // see peers' h writes
    }
}

// ---------- host launcher ----------
extern "C" void kernel_launch(void* const* d_in, const int* in_sizes, int n_in,
                              void* d_out, int out_size, void* d_ws, size_t ws_size,
                              hipStream_t stream) {
    const float* x    = (const float*)d_in[0];
    const float* h0   = (const float*)d_in[1];
    const float* cst  = (const float*)d_in[2];
    const float* kern = (const float*)d_in[3];
    const float* rker = (const float*)d_in[4];
    float* out = (float*)d_out;

    char* ws = (char*)d_ws;
    const size_t OFF_KT  = 0;
    const size_t OFF_RT  = (size_t)2 * 1024 * 1024;
    const size_t OFF_HB  = (size_t)4 * 1024 * 1024;
    const size_t OFF_CNT = OFF_HB + (size_t)2 * B_ * U_ * sizeof(unsigned short);
    const size_t OFF_XBF = (size_t)8 * 1024 * 1024;
    const size_t XBF_BYTES = (size_t)B_ * T_ * D_ * sizeof(unsigned short);

    unsigned short* Kt   = (unsigned short*)(ws + OFF_KT);
    unsigned short* Rt   = (unsigned short*)(ws + OFF_RT);
    unsigned short* hbuf = (unsigned short*)(ws + OFF_HB);
    unsigned*       cnt  = (unsigned*)(ws + OFF_CNT);
    unsigned short* xbf  = (unsigned short*)(ws + OFF_XBF);

    convert_weights<<<(D_ * U_) / 256, 256, 0, stream>>>(kern, rker, Kt, Rt);
    init_h<<<(B_ * U_ + 255) / 256, 256, 0, stream>>>(h0, hbuf, cnt);

    dim3 g1((B_ * T_) / 128, U_ / 64);
    if (ws_size >= OFF_XBF + XBF_BYTES) {
        const size_t n4 = (size_t)B_ * T_ * D_ / 4;
        convert_x<<<(unsigned)(n4 / 256), 256, 0, stream>>>(x, xbf);
        xw_gemm<true><<<g1, 128, 0, stream>>>(x, xbf, Kt, cst, out);
    } else {
        xw_gemm<false><<<g1, 128, 0, stream>>>(x, xbf, Kt, cst, out);
    }

    scan_kernel<<<NWG2, 64, 0, stream>>>(out, Rt, hbuf, cnt);
}
